// Cosine_PredictingModule_88622355186223
// MI455X (gfx1250) — matrix operand order repaired and checked by reading, not compiled, not run on hardware
//
#include <hip/hip_runtime.h>
#include <stdint.h>

// ---------------------------------------------------------------------------
// CDNA5 (gfx1250, wave32) edge-MLP, transposed formulation:
//   D1 = W1 * X^T  (X = [h,t] per edge), bias+cos column folded in as an
//   extra K-step;  D2 = W2 * relu(D1) with b2 folded in;  layer 3 in VALU.
// A-fragments = weights (LDS, TDM-staged), B-fragments = edge data (lane =
// edge column). One wave32 owns each 16-edge tile; no block barriers in the
// hot loop. sched_barriers bound the scheduler window per M-tile so the
// compiler cannot hoist-and-spill all A fragments.
// ---------------------------------------------------------------------------

typedef __attribute__((ext_vector_type(16))) _Float16 v16h;
typedef __attribute__((ext_vector_type(8)))  float    v8f;
typedef __attribute__((ext_vector_type(2)))  _Float16 h2;
typedef unsigned int v4u __attribute__((ext_vector_type(4)));
typedef int          v8i __attribute__((ext_vector_type(8)));
typedef int          v4i __attribute__((ext_vector_type(4)));

#if __has_builtin(__builtin_amdgcn_sched_barrier)
#define SCHED_FENCE() __builtin_amdgcn_sched_barrier(0)
#else
#define SCHED_FENCE()
#endif

#define W1_STRIDE 132          // dwords per W1 row (128 f16-pairs + 4 pad)
#define W2_STRIDE 68           // dwords per W2 row (64 f16-pairs + 4 pad)
#define W1_DW (128 * W1_STRIDE)        // 16896
#define W2_DW (32  * W2_STRIDE)        // 2176
#define E1_OFF (W1_DW + W2_DW)         // pack(b1[row], W1c[row]) : 128 dwords
#define E2_OFF (E1_OFF + 128)          // pack(b2[row], 0)        : 32 dwords
#define LDS_DW (E2_OFF + 32)           // 19232 dwords = 76928 bytes

union AFrag {
  v16h     v;
  uint32_t u[8];
  uint4    q[2];
};

static __device__ __forceinline__ uint32_t pk2(float a, float b) {
  union { h2 p; uint32_t u; } x;
  x.p[0] = (_Float16)a;
  x.p[1] = (_Float16)b;
  return x.u;
}

static __device__ __forceinline__ void acc3(const float4& h, const float4& t,
                                            float& dot, float& hh, float& tt) {
  dot += h.x * t.x + h.y * t.y + h.z * t.z + h.w * t.w;
  hh  += h.x * h.x + h.y * h.y + h.z * h.z + h.w * h.w;
  tt  += t.x * t.x + t.y * t.y + t.z * t.z + t.w * t.w;
}

// ---------------------------------------------------------------------------
// Prep: pack weight rows as f16 K-pairs (A-fragment layout), plus the folded
// bias/cos columns:  e1[row] = pack(b1[row], W1[row][256]),
//                    e2[row] = pack(b2[row], 0).
// ---------------------------------------------------------------------------
__global__ void prep_weights_kernel(const float* __restrict__ W1,
                                    const float* __restrict__ b1,
                                    const float* __restrict__ W2,
                                    const float* __restrict__ b2,
                                    uint32_t* __restrict__ wsbuf) {
  int t = blockIdx.x * blockDim.x + threadIdx.x;
  int stride = gridDim.x * blockDim.x;
  for (int i = t; i < 128 * 128; i += stride) {
    int row = i >> 7, q = i & 127;
    const float* r = W1 + row * 257 + 2 * q;
    wsbuf[row * W1_STRIDE + q] = pk2(r[0], r[1]);
  }
  for (int i = t; i < 32 * 64; i += stride) {
    int row = i >> 6, q = i & 63;
    const float* r = W2 + row * 128 + 2 * q;
    wsbuf[W1_DW + row * W2_STRIDE + q] = pk2(r[0], r[1]);
  }
  for (int i = t; i < 128; i += stride)
    wsbuf[E1_OFF + i] = pk2(b1[i], W1[i * 257 + 256]);
  for (int i = t; i < 32; i += stride)
    wsbuf[E2_OFF + i] = pk2(b2[i], 0.f);
}

// ---------------------------------------------------------------------------
// Main kernel: 256 threads = 8 waves; each wave owns independent 16-edge tiles.
// ---------------------------------------------------------------------------
__global__ void __launch_bounds__(256) edge_mlp_kernel(
    const float* __restrict__ h_user, const float* __restrict__ h_item,
    const int* __restrict__ src, const int* __restrict__ dst,
    const float* __restrict__ W3g, const float* __restrict__ b3g,
    const uint32_t* __restrict__ wsbuf,
    float* __restrict__ out, int E) {
  extern __shared__ uint32_t lds[];
  const uint32_t* lw1   = lds;
  const uint32_t* lw2   = lds + W1_DW;
  const uint32_t* lw_e1 = lds + E1_OFF;
  const uint32_t* lw_e2 = lds + E2_OFF;

  const int lane = threadIdx.x & 31;
  const int wv   = threadIdx.x >> 5;
  const int hv   = lane >> 4;   // lane half (K sub-range per ISA layouts)
  const int lm   = lane & 15;   // edge column (B/C/D) or weight row (A)

  // ---- Stage packed weights into LDS via Tensor Data Mover ----------------
#if __has_builtin(__builtin_amdgcn_tensor_load_to_lds) && \
    __has_builtin(__builtin_amdgcn_s_wait_tensorcnt)
  if (wv == 0) {
    const uint64_t ga = (uint64_t)(uintptr_t)wsbuf;
    const uint32_t elems = (uint32_t)(LDS_DW / 2);   // 8-byte elements
    v4u g0;
    g0[0] = 1u;                                   // count=1, user mode
    g0[1] = 0u;                                   // lds_addr = 0
    g0[2] = (uint32_t)ga;                         // global_addr[31:0]
    g0[3] = (uint32_t)((ga >> 32) & 0x01FFFFFFu)  // global_addr[56:32]
            | 0x80000000u;                        // type=2 ("image")
    v8i g1;
    g1[0] = (int)0x00030000u;                     // data_size=3 (8B), mask=0
    g1[1] = (int)((elems & 0xFFFFu) << 16);       // tensor_dim0[15:0] @63:48
    g1[2] = (int)((elems >> 16) | (1u << 16));    // tdim0[31:16], tensor_dim1=1
    g1[3] = (int)(elems << 16);                   // tile_dim0 @127:112
    g1[4] = 1;                                    // tile_dim1=1, tile_dim2=0
    g1[5] = (int)elems;                           // tensor_dim0_stride[31:0]
    g1[6] = 0;
    g1[7] = 0;
    v4i gz = {0, 0, 0, 0};
    v8i gz8 = {0, 0, 0, 0, 0, 0, 0, 0};
    __builtin_amdgcn_tensor_load_to_lds(g0, g1, gz, gz, gz8, 0);
    __builtin_amdgcn_s_wait_tensorcnt(0);
  }
  __syncthreads();
#else
  {
    const uint4* s4 = (const uint4*)wsbuf;
    uint4* d4 = (uint4*)lds;
    for (int i = threadIdx.x; i < LDS_DW / 4; i += blockDim.x) d4[i] = s4[i];
  }
  __syncthreads();
#endif

  // ---- Per-lane tile-invariant constants (hoisted out of the hot loop) ----
  float w3v[2][8];
#pragma unroll
  for (int m2 = 0; m2 < 2; m2++)
#pragma unroll
    for (int r = 0; r < 8; r++) w3v[m2][r] = W3g[16 * m2 + 8 * hv + r];
  const float b3 = b3g[0];
  // Folded bias/cos A-columns, loaded once and half-lane selected up front:
  uint32_t e1sel[8];
#pragma unroll
  for (int m = 0; m < 8; m++) {
    const uint32_t t = lw_e1[16 * m + lm];   // unconditional load, no branch
    e1sel[m] = hv ? 0u : t;
  }
  uint32_t e2sel[2];
#pragma unroll
  for (int m2 = 0; m2 < 2; m2++) {
    const uint32_t t = lw_e2[16 * m2 + lm];
    e2sel[m2] = hv ? 0u : t;
  }
  const uint32_t onesel = hv ? 0u : 0x00003C00u;   // pack(1.0h, 0)

  const int tilesTotal = (E + 15) >> 4;
  for (int tile = blockIdx.x * 8 + wv; tile < tilesTotal; tile += gridDim.x * 8) {
    const long long base = (long long)tile * 16;
    long long e = base + lm;
    if (e >= E) e = (long long)E - 1;
    const float* hrow = h_user + (size_t)src[e] * 128;
    const float* trow = h_item + (size_t)dst[e] * 128;

    // ---- Gather edge data as B-fragments (lane = edge column) ------------
    // B 32x16, contiguous K order used here (same as A): lanes 0-15 hold K=0..15 (u[v] = pair {2v,2v+1}),
    // lanes 16-31 hold K=16..31 -> this lane loads 16 contiguous floats
    // starting at feature 32k + 16*hv.  xb[0..3]=h, xb[4..7]=t.
    AFrag xb[8];
    float dot = 0.f, hh = 0.f, tt = 0.f;
#pragma unroll
    for (int k = 0; k < 4; k++) {
      const float* hp = hrow + 32 * k + 16 * hv;
      const float* tp = trow + 32 * k + 16 * hv;
      float4 q0 = *(const float4*)(hp);
      float4 q1 = *(const float4*)(hp + 4);
      float4 q2 = *(const float4*)(hp + 8);
      float4 q3 = *(const float4*)(hp + 12);
      float4 r0 = *(const float4*)(tp);
      float4 r1 = *(const float4*)(tp + 4);
      float4 r2 = *(const float4*)(tp + 8);
      float4 r3 = *(const float4*)(tp + 12);
      acc3(q0, r0, dot, hh, tt);
      acc3(q1, r1, dot, hh, tt);
      acc3(q2, r2, dot, hh, tt);
      acc3(q3, r3, dot, hh, tt);
      xb[k].u[0] = pk2(q0.x, q0.y);  xb[k].u[1] = pk2(q0.z, q0.w);
      xb[k].u[2] = pk2(q1.x, q1.y);  xb[k].u[3] = pk2(q1.z, q1.w);
      xb[k].u[4] = pk2(q2.x, q2.y);  xb[k].u[5] = pk2(q2.z, q2.w);
      xb[k].u[6] = pk2(q3.x, q3.y);  xb[k].u[7] = pk2(q3.z, q3.w);
      xb[k + 4].u[0] = pk2(r0.x, r0.y);  xb[k + 4].u[1] = pk2(r0.z, r0.w);
      xb[k + 4].u[2] = pk2(r1.x, r1.y);  xb[k + 4].u[3] = pk2(r1.z, r1.w);
      xb[k + 4].u[4] = pk2(r2.x, r2.y);  xb[k + 4].u[5] = pk2(r2.z, r2.w);
      xb[k + 4].u[6] = pk2(r3.x, r3.y);  xb[k + 4].u[7] = pk2(r3.z, r3.w);
    }
    // Lane pair (L, L+16) holds complementary feature halves of edge lm.
    dot += __shfl_xor(dot, 16, 32);
    hh  += __shfl_xor(hh, 16, 32);
    tt  += __shfl_xor(tt, 16, 32);
    const float cosv = dot * __builtin_amdgcn_rsqf(fmaxf(hh, 1e-24f))
                           * __builtin_amdgcn_rsqf(fmaxf(tt, 1e-24f));
    // Folded bias+cos column: B_extra[K=0]=1, [K=1]=cos  (lane half 0 only).
    const uint32_t one_cos = pk2(1.0f, cosv);
    AFrag bX;
    bX.u[0] = hv ? 0u : one_cos;
#pragma unroll
    for (int v = 1; v < 8; v++) bX.u[v] = 0u;

    // ---- Layer 1: D1 = W1 * X^T, 8 M-tiles x (8+1) K-steps ----------------
    // D layout: lane = edge (lm), vgpr r -> feature 16m + 8*hv + r.
    // Keep relu'd outputs as packed f16 pairs p[m][vv] = (feat 2vv, 2vv+1).
    uint32_t p[8][4];
    const uint32_t* abase1 = lw1 + (size_t)lm * W1_STRIDE + 8 * hv;
#pragma unroll
    for (int m = 0; m < 8; m++) {
      const uint32_t* arow = abase1 + (size_t)(16 * m) * W1_STRIDE;
      v8f c = {};
#pragma unroll
      for (int k = 0; k < 8; k++) {
        AFrag aW;   // A 16x32 map: q = 16k + 8hv + v
        aW.q[0] = *(const uint4*)(arow + 16 * k);
        aW.q[1] = *(const uint4*)(arow + 16 * k + 4);
        c = __builtin_amdgcn_wmma_f32_16x16x32_f16(false, aW.v, false, xb[k].v,
                                                   (short)0, c, false, false);
      }
      AFrag aX;   // bias/cos column: A[K=0]=b1[row], [K=1]=W1c[row]
      aX.u[0] = e1sel[m];
#pragma unroll
      for (int v = 1; v < 8; v++) aX.u[v] = 0u;
      c = __builtin_amdgcn_wmma_f32_16x16x32_f16(false, aX.v, false, bX.v,
                                                 (short)0, c, false, false);
#pragma unroll
      for (int vv = 0; vv < 4; vv++)
        p[m][vv] = pk2(fmaxf(c[2 * vv], 0.f), fmaxf(c[2 * vv + 1], 0.f));
      // Stop the scheduler from hoisting the next M-tile's A loads (and
      // spilling): one M-tile of LDS loads + WMMAs per window is enough.
      SCHED_FENCE();
    }

    // ---- Layer 2: D2 = W2 * relu(D1), 2 M-tiles x (4+1) K-steps -----------
    // B2 fragment for K-step k2: features 32k2 + 16hv + {0..15}.  Feature
    // f lives on lane-half (f&15)>=8, pack index m' = f>>4 -> mix own packs
    // p[2k2+hv] with partner's (shfl_xor 16).
    float s = 0.f;
    const uint32_t* abase2 = lw2 + (size_t)lm * W2_STRIDE + 8 * hv;
#pragma unroll
    for (int m2 = 0; m2 < 2; m2++) {
      const uint32_t* arow = abase2 + (size_t)(16 * m2) * W2_STRIDE;
      v8f c = {};
#pragma unroll
      for (int k2 = 0; k2 < 4; k2++) {
        AFrag aW;
        aW.q[0] = *(const uint4*)(arow + 16 * k2);
        aW.q[1] = *(const uint4*)(arow + 16 * k2 + 4);
        AFrag bY;
        const int m0 = 2 * k2, m1 = 2 * k2 + 1;
#pragma unroll
        for (int vv = 0; vv < 4; vv++) {
          const uint32_t pe0 = (uint32_t)__shfl_xor((int)p[m0][vv], 16, 32);
          const uint32_t pe1 = (uint32_t)__shfl_xor((int)p[m1][vv], 16, 32);
          bY.u[vv]     = hv ? pe1 : p[m0][vv];
          bY.u[vv + 4] = hv ? p[m1][vv] : pe0;
        }
        c = __builtin_amdgcn_wmma_f32_16x16x32_f16(false, aW.v, false, bY.v,
                                                   (short)0, c, false, false);
      }
      AFrag aX2;  // folded b2 column
      aX2.u[0] = e2sel[m2];
#pragma unroll
      for (int v = 1; v < 8; v++) aX2.u[v] = 0u;
      AFrag bX2;
      bX2.u[0] = onesel;
#pragma unroll
      for (int v = 1; v < 8; v++) bX2.u[v] = 0u;
      c = __builtin_amdgcn_wmma_f32_16x16x32_f16(false, aX2.v, false, bX2.v,
                                                 (short)0, c, false, false);
      // Layer 3 partial: s += relu(c[r]) * W3[16m2 + 8hv + r]
#pragma unroll
      for (int r = 0; r < 8; r++) s += fmaxf(c[r], 0.f) * w3v[m2][r];
      SCHED_FENCE();
    }
    // Partner half holds the other 16 features of this edge.
    s += __shfl_xor(s, 16, 32);
    if (hv == 0) {
      const long long o = base + lm;
      if (o < (long long)E)
        out[o] = 1.f / (1.f + __expf(-(s + b3)));      // sigmoid
    }
  }
}

// ---------------------------------------------------------------------------
extern "C" void kernel_launch(void* const* d_in, const int* in_sizes, int n_in,
                              void* d_out, int out_size, void* d_ws, size_t ws_size,
                              hipStream_t stream) {
  const float* h_user = (const float*)d_in[0];
  const float* h_item = (const float*)d_in[1];
  const int*   src    = (const int*)d_in[2];
  const int*   dst    = (const int*)d_in[3];
  const float* W1     = (const float*)d_in[4];
  const float* b1     = (const float*)d_in[5];
  const float* W2     = (const float*)d_in[6];
  const float* b2     = (const float*)d_in[7];
  const float* W3     = (const float*)d_in[8];
  const float* b3     = (const float*)d_in[9];
  float* out = (float*)d_out;
  const int E = in_sizes[2];
  uint32_t* wsbuf = (uint32_t*)d_ws;

  hipLaunchKernelGGL(prep_weights_kernel, dim3(64), dim3(256), 0, stream,
                     W1, b1, W2, b2, wsbuf);

  const size_t ldsBytes = (size_t)LDS_DW * sizeof(uint32_t);  // ~76.9 KB / block
  int tiles = (E + 15) >> 4;
  int blocks = (tiles + 7) / 8;
  if (blocks > 1024) blocks = 1024;
  if (blocks < 1) blocks = 1;
  hipLaunchKernelGGL(edge_mlp_kernel, dim3(blocks), dim3(256), ldsBytes, stream,
                     h_user, h_item, src, dst, W3, b3, wsbuf, out, E);
}
